// URNNCell_91122026152843
// MI455X (gfx1250) — matrix operand order repaired and checked by reading, not compiled, not run on hardware
//
#include <hip/hip_runtime.h>
#include <hip/hip_bf16.h>

// ---------------------------------------------------------------------------
// URNNCell forward, dead-code-eliminated: out = modrelu(inputs @ w_ih^T, b_h)
//   B=16384, I=512, H=512.
//   - GEMM via fp32-accurate bf16x3 (hi*hi + lo*hi + hi*lo) on
//     v_wmma_f32_16x16x32_bf16
//   - Operand staging via Tensor Data Mover (tensor_load_to_lds) into
//     double-buffered LDS, TDM pad feature used to de-conflict LDS banks,
//     synchronized with s_wait_tensorcnt + workgroup barrier.
//   - Fused modReLU epilogue (no workspace, single pass over the output).
// ---------------------------------------------------------------------------

typedef __attribute__((ext_vector_type(16))) __bf16       v16bf;
typedef __attribute__((ext_vector_type(8)))  float        v8f;
typedef __attribute__((ext_vector_type(4)))  float        f4;
typedef __attribute__((ext_vector_type(4)))  unsigned int u32x4;
typedef __attribute__((ext_vector_type(8)))  int          i32x8;
typedef __attribute__((ext_vector_type(4)))  int          i32x4;

#define Bsz 16384
#define Isz 512
#define Hsz 512
#define KC  32                 // K elements per pipeline chunk
#define AST 36                 // LDS row stride in dwords: 32 data + 4 TDM pad
#define NCHUNK (Isz / KC)      // 16

static __device__ __forceinline__ v8f wmma_bf16(v16bf a, v16bf b, v8f c) {
  return __builtin_amdgcn_wmma_f32_16x16x32_bf16(
      false, a, false, b, (short)0, c, false, false);
}

// ---- Tensor Data Mover: 2D tile (tile_d0 x tile_d1) of f32, with LDS pad of
// ---- 4 dwords every 32 dwords (one tile row) -> padded row stride 36 dwords.
static __device__ __forceinline__ void tdm_load_2d(
    unsigned lds_byte_addr, const float* gaddr,
    unsigned tensor_d0, unsigned tensor_d1,
    unsigned tile_d0, unsigned tile_d1, unsigned stride0) {
  unsigned long long ga = (unsigned long long)(uintptr_t)gaddr;

  u32x4 g0;
  g0[0] = 1u;                                          // count=1, user D#
  g0[1] = lds_byte_addr;                               // lds_addr
  g0[2] = (unsigned)(ga & 0xffffffffu);                // global_addr[31:0]
  g0[3] = (unsigned)((ga >> 32) & 0x1ffffffu)          // global_addr[56:32]
        | (2u << 30);                                  // type = 2 ("image")

  i32x8 g1;
  g1[0] = (int)((2u << 16)                             // data_size = 4 bytes
              | (1u << 20)                             // pad_enable
              | (4u << 22)                             // pad_interval: 32 DW
              | (3u << 25));                           // pad_amount:   4 DW
  g1[1] = (int)((tensor_d0 & 0xffffu) << 16);          // abar=0 | td0[15:0]
  g1[2] = (int)((tensor_d0 >> 16) | ((tensor_d1 & 0xffffu) << 16));
  g1[3] = (int)((tensor_d1 >> 16) | (tile_d0 << 16));  // td1[31:16] | tile_d0
  g1[4] = (int)(tile_d1 & 0xffffu);                    // tile_d1 | tile_d2=0
  g1[5] = (int)stride0;                                // dim0_stride[31:0]
  g1[6] = 0;                                           // stride hi | d1s lo
  g1[7] = 0;

  i32x4 z4 = {0, 0, 0, 0};                             // 2D: groups 2/3 unused
  i32x8 z8 = {0, 0, 0, 0, 0, 0, 0, 0};                 // trailing group unused
  __builtin_amdgcn_tensor_load_to_lds(g0, g1, z4, z4, z8, 0);
}

// ---- bf16 hi/lo split of 16 fp32 values ----------------------------------
static __device__ __forceinline__ void split16(const float* __restrict__ f,
                                               v16bf& hi, v16bf& lo) {
#pragma unroll
  for (int i = 0; i < 16; ++i) {
    float  x = f[i];
    __bf16 h = (__bf16)x;
    hi[i] = h;
    lo[i] = (__bf16)(x - (float)h);
  }
}

// A fragment (16-bit A 16x32 layout): lane holds K = kb..kb+7 and kb+16..kb+23
// (kb = 8*khalf). p = &row[8*khalf] inside a 32-dword LDS row.
static __device__ __forceinline__ void load_a(const float* p,
                                              v16bf& hi, v16bf& lo) {
  float f[16];
  *(f4*)(f + 0)  = *(const f4*)(p + 0);
  *(f4*)(f + 4)  = *(const f4*)(p + 4);
  *(f4*)(f + 8)  = *(const f4*)(p + 16);
  *(f4*)(f + 12) = *(const f4*)(p + 20);
  split16(f, hi, lo);
}

// B fragment (16-bit B 32x16 layout): lane holds K = kb..kb+7 and kb+16..kb+23
// (kb = 8*khalf) of column n = W-row n in LDS.
static __device__ __forceinline__ void load_b(const float* p,
                                              v16bf& hi, v16bf& lo) {
  float f[16];
  *(f4*)(f + 0)  = *(const f4*)(p + 0);
  *(f4*)(f + 4)  = *(const f4*)(p + 4);
  *(f4*)(f + 8)  = *(const f4*)(p + 16);
  *(f4*)(f + 12) = *(const f4*)(p + 20);
  split16(f, hi, lo);
}

__global__ __launch_bounds__(256)
void urnn_fused_tdm(const float* __restrict__ A,   // inputs  (16384, 512)
                    const float* __restrict__ W,   // w_ih    (1024, 512)
                    const float* __restrict__ bh,  // b_h     (512)
                    float* __restrict__ out) {     // output  (16384, 1024)
  // Double-buffered operand tiles (f32), TDM-padded rows (36 dwords):
  //   sA: 128 rows x 32 k   sB: [re/im] 64 rows x 32 k      => 72 KB total
  __shared__ float sA[2][128 * AST];
  __shared__ float sB[2][2][64 * AST];

  const int lane   = threadIdx.x & 31;
  const int wid    = threadIdx.x >> 5;
  const int lane16 = lane & 15;
  const int khalf  = lane >> 4;
  const int wrow   = wid & 3;           // 4 waves along M (32 rows each)
  const int wcol   = wid >> 2;          // 2 waves along N (32 cols each)

  const int mblk = blockIdx.x * 128;    // block row base
  const int nblk = blockIdx.y * 64;     // block complex-column base

  v8f acc[2][2][2] = {};                // [re/im][mi][ni]

  auto issue_chunk = [&](int buf, int k0) {
    tdm_load_2d((unsigned)(uintptr_t)&sA[buf][0],
                A + (long)mblk * Isz + k0,
                Isz, Bsz, KC, 128, Isz);
    tdm_load_2d((unsigned)(uintptr_t)&sB[buf][0][0],
                W + (long)nblk * Isz + k0,
                Isz, 2 * Hsz, KC, 64, Isz);
    tdm_load_2d((unsigned)(uintptr_t)&sB[buf][1][0],
                W + (long)(nblk + Hsz) * Isz + k0,
                Isz, 2 * Hsz, KC, 64, Isz);
  };

  // Prologue: stage chunk 0.
  if (wid == 0) {
    issue_chunk(0, 0);
    __builtin_amdgcn_s_wait_tensorcnt(0);
  }
  __syncthreads();

  for (int kc = 0; kc < NCHUNK; ++kc) {
    const int buf = kc & 1;

    // Kick the DMA for the next chunk while we compute this one.
    if (wid == 0 && kc + 1 < NCHUNK) issue_chunk(buf ^ 1, (kc + 1) * KC);

    // ---- fragments from LDS (ds_load_b128, conflict-free via 36-DW stride)
    v16bf ahi[2], alo[2];
    const float* ap = &sA[buf][(wrow * 32 + lane16) * AST + 8 * khalf];
    load_a(ap,            ahi[0], alo[0]);
    load_a(ap + 16 * AST, ahi[1], alo[1]);

    v16bf bhi[2][2], blo[2][2];
#pragma unroll
    for (int h = 0; h < 2; ++h)
#pragma unroll
      for (int ni = 0; ni < 2; ++ni) {
        const float* bp =
            &sB[buf][h][(wcol * 32 + 16 * ni + lane16) * AST + 8 * khalf];
        load_b(bp, bhi[h][ni], blo[h][ni]);
      }

    // ---- bf16x3 WMMA: 24 matrix ops per chunk
#pragma unroll
    for (int h = 0; h < 2; ++h)
#pragma unroll
      for (int mi = 0; mi < 2; ++mi)
#pragma unroll
        for (int ni = 0; ni < 2; ++ni) {
          v8f c = acc[h][mi][ni];
          c = wmma_bf16(ahi[mi], bhi[h][ni], c);  // hi*hi
          c = wmma_bf16(alo[mi], bhi[h][ni], c);  // lo*hi
          c = wmma_bf16(ahi[mi], blo[h][ni], c);  // hi*lo
          acc[h][mi][ni] = c;
        }

    // Next buffer must be resident before anyone reads it.
    if (wid == 0 && kc + 1 < NCHUNK) __builtin_amdgcn_s_wait_tensorcnt(0);
    __syncthreads();
  }

  // ---- Fused modReLU epilogue: scale = relu(|z|+b)/(|z|+1e-6) -------------
#pragma unroll
  for (int ni = 0; ni < 2; ++ni) {
    const int   j = nblk + wcol * 32 + 16 * ni + lane16;
    const float b = bh[j];
#pragma unroll
    for (int mi = 0; mi < 2; ++mi) {
#pragma unroll
      for (int r = 0; r < 8; ++r) {
        const int row = mblk + wrow * 32 + 16 * mi + r + 8 * khalf;
        float re  = acc[0][mi][ni][r];
        float im  = acc[1][mi][ni][r];
        float nrm = sqrtf(re * re + im * im);
        float sc  = fmaxf(nrm + b, 0.0f) / (nrm + 1e-6f);
        float* o  = out + (long)row * (2 * Hsz) + j;
        o[0]   = re * sc;
        o[Hsz] = im * sc;
      }
    }
  }
}

extern "C" void kernel_launch(void* const* d_in, const int* in_sizes, int n_in,
                              void* d_out, int out_size, void* d_ws, size_t ws_size,
                              hipStream_t stream) {
  // setup_inputs order: inputs, state, w_ih, b_h, d1, d2, d3, re1, im1, re2,
  // im2, perm.  Everything except inputs/w_ih/b_h is dead in the reference
  // (preact is never used).
  const float* inputs = (const float*)d_in[0];
  const float* w_ih   = (const float*)d_in[2];
  const float* b_h    = (const float*)d_in[3];
  float*       out    = (float*)d_out;

  dim3 grid(Bsz / 128, Hsz / 64);  // 128 x 8 blocks
  dim3 block(256);                 // 8 waves (wave32)
  urnn_fused_tdm<<<grid, block, 0, stream>>>(inputs, w_ih, b_h, out);

  (void)in_sizes; (void)n_in; (void)out_size; (void)d_ws; (void)ws_size;
}
